// TGSL_90469191123535
// MI455X (gfx1250) — compile-verified, matrix operand order checked by reading, not run on hardware
//
#include <hip/hip_runtime.h>
#include <hip/hip_bf16.h>
#include <math.h>

#define N_NODES 200000
#define EDIM    128
#define BSZ     4096
#define HNB     20
#define NNEG    5
#define HID     60
#define HIDP    64
#define SIGMA   1.0f
#define MAXT    1.0f
#define INDEXC  4096.0f

typedef __attribute__((ext_vector_type(16))) __bf16 v16bf;
typedef __attribute__((ext_vector_type(8)))  float  v8f;

__device__ __forceinline__ unsigned short f32_to_bf16(float f) {
  unsigned int u = __float_as_uint(f);
  u += 0x7FFFu + ((u >> 16) & 1u);   // round-to-nearest-even
  return (unsigned short)(u >> 16);
}

// Fragment K-mapping from CDNA5 ISA 7.12.2 (16-bit A 16x32):
// lanes 0-15 hold K {0..7,16..23}(+k0), lanes 16-31 hold K {8..15,24..31}(+k0).
// Elements 0..7 and 8..15 are K-contiguous -> two 16B LDS loads.
__device__ __forceinline__ v16bf frag_u16(const unsigned short* rowk0, int g) {
  union { v16bf v; uint4 q[2]; } r;
  const uint4* p = (const uint4*)(rowk0 + (g ? 8 : 0));
  r.q[0] = p[0];   // K = goff + 0..7
  r.q[1] = p[2];   // K = goff + 16..23
  return r.v;
}

__device__ __forceinline__ v16bf frag_f32(const float* rowk0, int g) {
  union { v16bf v; unsigned short u[16]; } r;
  const float* p = rowk0 + (g ? 8 : 0);
#pragma unroll
  for (int i = 0; i < 8; ++i) {
    r.u[i]     = f32_to_bf16(p[i]);
    r.u[8 + i] = f32_to_bf16(p[16 + i]);
  }
  return r.v;
}

__device__ __forceinline__ v8f wmma_bf16(v16bf a, v16bf b, v8f c) {
  return __builtin_amdgcn_wmma_f32_16x16x32_bf16(
      false, a, false, b, (short)0, c, false, false);
}

// Async global->LDS copy of 16 bytes (GVS mode: SGPR64 base + VGPR32 offset).
// ldsByteAddr is the low-32 truncation of the generic LDS pointer (ISA 10.2).
__device__ __forceinline__ void async_copy_b128(unsigned ldsByteAddr,
                                                unsigned globOff,
                                                const void* base) {
  asm volatile("global_load_async_to_lds_b128 %0, %1, %2"
               :: "v"(ldsByteAddr), "v"(globOff), "s"(base) : "memory");
}
__device__ __forceinline__ void async_wait0() {
  asm volatile("s_wait_asynccnt 0" ::: "memory");
}
__device__ __forceinline__ unsigned lds_addr(const void* p) {
  return (unsigned)(size_t)p;
}

// ---------------------------------------------------------------------------
// Kernel 1: node_emb = relu(X@W1+b1)@W2+b2   (bf16 WMMA, f32 accumulate)
// One block = 256 threads = 8 waves, 128 node rows (16 per wave).
// Input tile staged via GLOBAL_LOAD_ASYNC_TO_LDS_B128 (ASYNCcnt).
// ---------------------------------------------------------------------------
__global__ __launch_bounds__(256) void tgsl_mlp_kernel(
    const float* __restrict__ X,  const float* __restrict__ W1,
    const float* __restrict__ b1, const float* __restrict__ W2,
    const float* __restrict__ b2, float* __restrict__ emb)
{
  __shared__ alignas(16) float sAf[128 * 128];             // 64 KB raw f32 tile
  __shared__ alignas(16) unsigned short sW1t[HIDP * 128];  // [j][k] transposed
  __shared__ alignas(16) unsigned short sW2t[128 * HIDP];  // [e][k] transposed
  __shared__ alignas(16) unsigned short sH[8 * 16 * HIDP]; // per-wave hidden
  __shared__ float sb1[HIDP];
  __shared__ float sb2[128];

  const int tid = threadIdx.x;
  const long row0 = (long)blockIdx.x * 128;

  for (int i = tid; i < HIDP * 128; i += 256) {
    int j = i >> 7, k = i & 127;
    sW1t[i] = f32_to_bf16(j < HID ? W1[k * HID + j] : 0.0f);
  }
  for (int i = tid; i < 128 * HIDP; i += 256) {
    int e = i >> 6, k = i & 63;
    sW2t[i] = f32_to_bf16(k < HID ? W2[k * 128 + e] : 0.0f);
  }
  if (tid < HIDP) sb1[tid] = (tid < HID) ? b1[tid] : 0.0f;
  if (tid < 128)  sb2[tid] = b2[tid];

  if (row0 + 128 <= N_NODES) {
    // async DMA: 64KB contiguous tile, 16B per thread per pass
    const float* Xb = X + row0 * EDIM;
    const unsigned lbase = lds_addr(sAf);
#pragma unroll
    for (int it = 0; it < 16; ++it) {
      unsigned off = (unsigned)(tid * 16 + it * 4096);
      async_copy_b128(lbase + off, off, Xb);
    }
    async_wait0();
  } else {
    for (int i = tid; i < 128 * 128; i += 256) {
      int r = i >> 7, c = i & 127;
      long row = row0 + r;
      sAf[i] = (row < N_NODES) ? X[row * EDIM + c] : 0.0f;
    }
  }
  __syncthreads();

  const int wave = tid >> 5, lane = tid & 31;
  const int g = lane >> 4;           // lane group (0: lanes 0-15, 1: 16-31)
  const int n = lane & 15;           // A row / output column within tile
  const float* Aw = sAf + wave * 16 * 128;
  unsigned short* Hw = sH + wave * 16 * HIDP;

  // Layer 1: [16x128] x [128x64] -> 4 C tiles
  v8f acc1[4] = {};
#pragma unroll
  for (int ks = 0; ks < 4; ++ks) {
    v16bf af = frag_f32(Aw + n * 128 + ks * 32, g);
#pragma unroll
    for (int t = 0; t < 4; ++t) {
      v16bf bfv = frag_u16(sW1t + (t * 16 + n) * 128 + ks * 32, g);
      acc1[t] = wmma_bf16(af, bfv, acc1[t]);
    }
  }
  // bias + relu -> hidden in LDS (bf16), C layout: vgpr i = row i+8g, col n
#pragma unroll
  for (int t = 0; t < 4; ++t) {
    int cc = t * 16 + n;
#pragma unroll
    for (int i = 0; i < 8; ++i) {
      float h = acc1[t][i] + sb1[cc];
      Hw[(i + 8 * g) * HIDP + cc] = f32_to_bf16(h > 0.0f ? h : 0.0f);
    }
  }
  __syncthreads();

  // Layer 2: [16x64] x [64x128] -> 8 C tiles
  v8f acc2[8] = {};
#pragma unroll
  for (int ks = 0; ks < 2; ++ks) {
    v16bf af = frag_u16(Hw + n * HIDP + ks * 32, g);
#pragma unroll
    for (int t = 0; t < 8; ++t) {
      v16bf bfv = frag_u16(sW2t + (t * 16 + n) * HIDP + ks * 32, g);
      acc2[t] = wmma_bf16(af, bfv, acc2[t]);
    }
  }
#pragma unroll
  for (int t = 0; t < 8; ++t) {
    int cc = t * 16 + n;
#pragma unroll
    for (int i = 0; i < 8; ++i) {
      long row = row0 + wave * 16 + i + 8 * g;
      if (row < N_NODES) emb[row * EDIM + cc] = acc2[t][i] + sb2[cc];
    }
  }
}

// ---------------------------------------------------------------------------
// Kernel 2: per-edge attention / loss. One block per edge (4096 blocks).
// S tile rows: 0..19 = s_h_emb, 20 = pos0, 21 = pos1, 22..31 = 0
// T tile rows: 0..19 = t_h_emb, rest 0.  proj = tile @ W via bf16 WMMA.
// Row gathers (52 x 512B) done with one wave-wide async-b128 per row.
// ---------------------------------------------------------------------------
__global__ __launch_bounds__(256) void tgsl_edge_kernel(
    const float* __restrict__ emb,
    const int*   __restrict__ srcI, const int* __restrict__ dstI,
    const float* __restrict__ eT,
    const int*   __restrict__ shN,  const int* __restrict__ thN,
    const float* __restrict__ sTimes, const float* __restrict__ tTimes,
    const float* __restrict__ sMask,  const float* __restrict__ tMask,
    const int*   __restrict__ negSrcI, const int* __restrict__ negDstI,
    const float* __restrict__ Wm, const float* __restrict__ av,
    const float* __restrict__ dS_p, const float* __restrict__ dT_p,
    const float* __restrict__ gw, const float* __restrict__ gb,
    float* __restrict__ blockLoss, float* __restrict__ posOut)
{
  __shared__ alignas(16) unsigned short sWt[128 * 128]; // W transposed [n][k] bf16
  __shared__ alignas(16) float shS[32 * 128];
  __shared__ alignas(16) float shTb[32 * 128];
  __shared__ alignas(16) float prS[32 * 128];
  __shared__ alignas(16) float prT[32 * 128];
  __shared__ alignas(16) float ng0[NNEG * 128];
  __shared__ alignas(16) float ng1[NNEG * 128];
  __shared__ float scS[HNB], scT[HNB];
  __shared__ float simS[HNB], simT[HNB];
  __shared__ float dSa[HNB], dTa[HNB];
  __shared__ float attS[HNB], attT[HNB];
  __shared__ float wSs[HNB], wTt[HNB], aMs[HNB], aMt[HNB];
  __shared__ float pAs[HNB], pAt[HNB];
  __shared__ float nAs[HNB * NNEG], nAt[HNB * NNEG];
  __shared__ float nMuS[NNEG], nMuT[NNEG], nS[NNEG], nT[NNEG];
  __shared__ float nlS[NNEG], nlT[NNEG];
  __shared__ float gSv[128], gTv[128];
  __shared__ float sc[16];

  const int b = blockIdx.x;
  const int tid = threadIdx.x;
  const int wave = tid >> 5, lane = tid & 31;

  // W -> bf16 transposed (needs transpose+convert: stays on the VALU path)
  for (int i = tid; i < 128 * 128; i += 256) {
    int nn = i >> 7, k = i & 127;
    sWt[i] = f32_to_bf16(Wm[k * 128 + nn]);
  }

  // 52 async row gathers: each row = 512B = 32 lanes x 16B, one inst per row.
  {
    const unsigned laneB = (unsigned)lane * 16u;
    for (int t = wave; t < 52; t += 8) {
      int idx;
      const float* dstRow;
      if (t < 20)      { idx = shN[b * HNB + t];              dstRow = shS  + t * 128; }
      else if (t == 20){ idx = srcI[b];                       dstRow = shS  + 20 * 128; }
      else if (t == 21){ idx = dstI[b];                       dstRow = shS  + 21 * 128; }
      else if (t < 42) { idx = thN[b * HNB + (t - 22)];       dstRow = shTb + (t - 22) * 128; }
      else if (t < 47) { idx = negSrcI[b * NNEG + (t - 42)];  dstRow = ng0  + (t - 42) * 128; }
      else             { idx = negDstI[b * NNEG + (t - 47)];  dstRow = ng1  + (t - 47) * 128; }
      async_copy_b128(lds_addr(dstRow) + laneB,
                      (unsigned)idx * (EDIM * 4u) + laneB, emb);
    }
    // zero padding rows (plain ds stores, disjoint from async targets)
    for (int i = tid; i < 10 * 128; i += 256) shS[22 * 128 + i] = 0.0f;
    for (int i = tid; i < 12 * 128; i += 256) shTb[20 * 128 + i] = 0.0f;
    async_wait0();
  }
  __syncthreads();

  // Projection GEMMs: 32 output tiles (S:16, T:16), 4 per wave, K=128.
  {
    const int g = lane >> 4, n16 = lane & 15;
#pragma unroll
    for (int t4 = 0; t4 < 4; ++t4) {
      int tile = wave * 4 + t4;
      const float* Ab = (tile < 16) ? shS : shTb;
      float* Cb = (tile < 16) ? prS : prT;
      int u = tile & 15;
      int mt = u >> 3, nt = u & 7;
      v8f acc = {};
#pragma unroll
      for (int ks = 0; ks < 4; ++ks) {
        v16bf af  = frag_f32(Ab + (mt * 16 + n16) * 128 + ks * 32, g);
        v16bf bfv = frag_u16(sWt + (nt * 16 + n16) * 128 + ks * 32, g);
        acc = wmma_bf16(af, bfv, acc);
      }
#pragma unroll
      for (int i = 0; i < 8; ++i)
        Cb[(mt * 16 + i + 8 * g) * 128 + nt * 16 + n16] = acc[i];
    }
  }
  __syncthreads();

  const float* pos0 = shS + 20 * 128;
  const float* pos1 = shS + 21 * 128;

  // Phase A: dot products, squared distances (independent tasks)
  if (tid < 20) {
    float s = 0.f;
    for (int e = 0; e < 128; ++e) s += prS[tid * 128 + e] * av[128 + e];
    scS[tid] = s;
  } else if (tid < 40) {
    int h = tid - 20; float s = 0.f;
    for (int e = 0; e < 128; ++e) s += prT[h * 128 + e] * av[128 + e];
    scT[h] = s;
  } else if (tid == 40) {
    float s = 0.f; for (int e = 0; e < 128; ++e) s += prS[20 * 128 + e] * av[e];
    sc[0] = s;                                     // proj_s . a1
  } else if (tid == 41) {
    float s = 0.f; for (int e = 0; e < 128; ++e) s += prS[21 * 128 + e] * av[e];
    sc[1] = s;                                     // proj_t . a1
  } else if (tid == 42) {
    float s = 0.f;
    for (int e = 0; e < 128; ++e) { float d = pos0[e] - pos1[e]; s += d * d; }
    sc[2] = s;                                     // p_mu
  } else if (tid == 43) {
    float s = 0.f; for (int e = 0; e < 128; ++e) s += pos0[e] * pos1[e];
    sc[3] = s > 0.f ? s : 0.f;                     // pos
  } else if (tid == 44) {
    sc[4] = dS_p[srcI[b]];                         // ds
  } else if (tid == 45) {
    sc[5] = dT_p[dstI[b]];                         // dt
  } else if (tid < 51) {
    int nn = tid - 46; float s = 0.f;
    for (int e = 0; e < 128; ++e) s += pos0[e] * ng1[nn * 128 + e];
    nS[nn] = s > 0.f ? s : 0.f;                    // neg_s
  } else if (tid < 56) {
    int nn = tid - 51; float s = 0.f;
    for (int e = 0; e < 128; ++e) s += pos1[e] * ng0[nn * 128 + e];
    nT[nn] = s > 0.f ? s : 0.f;                    // neg_t
  } else if (tid < 76) {
    int h = tid - 56; float s = 0.f;
    for (int e = 0; e < 128; ++e) { float d = shS[h * 128 + e] - pos1[e]; s += d * d; }
    pAs[h] = s;
  } else if (tid < 96) {
    int h = tid - 76; float s = 0.f;
    for (int e = 0; e < 128; ++e) { float d = shTb[h * 128 + e] - pos0[e]; s += d * d; }
    pAt[h] = s;
  } else if (tid < 101) {
    int nn = tid - 96; float s = 0.f;
    for (int e = 0; e < 128; ++e) { float d = pos0[e] - ng1[nn * 128 + e]; s += d * d; }
    nMuS[nn] = s;
  } else if (tid < 106) {
    int nn = tid - 101; float s = 0.f;
    for (int e = 0; e < 128; ++e) { float d = pos1[e] - ng0[nn * 128 + e]; s += d * d; }
    nMuT[nn] = s;
  }
  if (tid < 200) {   // n_alpha_{s,t}[h][n]
    int side = tid / 100, task = tid % 100;
    int h = task / NNEG, nn = task % NNEG;
    const float* A_ = side ? shTb : shS;
    const float* Ng = side ? ng0 : ng1;
    float s = 0.f;
    for (int e = 0; e < 128; ++e) { float d = A_[h * 128 + e] - Ng[nn * 128 + e]; s += d * d; }
    if (side) nAt[h * NNEG + nn] = s; else nAs[h * NNEG + nn] = s;
  }
  __syncthreads();

  // Phase B1: time deltas + leaky-relu sims (ref quirk: sim_t uses ds too)
  if (tid < 20) {
    float et = eT[b] / MAXT;
    float d = fabsf(et - sTimes[b * HNB + tid] / MAXT);
    dSa[tid] = d;
    float x = expf(-sc[4] * d) * (sc[0] + scS[tid]);
    simS[tid] = x > 0.f ? x : 0.2f * x;
  } else if (tid < 40) {
    int h = tid - 20;
    float et = eT[b] / MAXT;
    float d = fabsf(et - tTimes[b * HNB + h] / MAXT);
    dTa[h] = d;
    float x = expf(-sc[4] * d) * (sc[1] + scT[h]);   // ds, per reference
    simT[h] = x > 0.f ? x : 0.2f * x;
  }
  __syncthreads();

  // Phase B2: unmasked softmax over H, mean time deltas
  if (tid == 0) {
    float mx = simS[0];
    for (int h = 1; h < HNB; ++h) mx = fmaxf(mx, simS[h]);
    float sum = 0.f;
    for (int h = 0; h < HNB; ++h) { float e = expf(simS[h] - mx); attS[h] = e; sum += e; }
    float inv = 1.f / sum;
    for (int h = 0; h < HNB; ++h) attS[h] *= inv;
    float md = 0.f;
    for (int h = 0; h < HNB; ++h) md += dSa[h];
    sc[6] = md / (float)HNB;
  } else if (tid == 1) {
    float mx = simT[0];
    for (int h = 1; h < HNB; ++h) mx = fmaxf(mx, simT[h]);
    float sum = 0.f;
    for (int h = 0; h < HNB; ++h) { float e = expf(simT[h] - mx); attT[h] = e; sum += e; }
    float inv = 1.f / sum;
    for (int h = 0; h < HNB; ++h) attT[h] *= inv;
    float md = 0.f;
    for (int h = 0; h < HNB; ++h) md += dTa[h];
    sc[7] = md / (float)HNB;
  }
  __syncthreads();

  // Phase B3: weighted attention factors
  if (tid < 20) {
    float e = expf(sc[4] * dSa[tid]);
    float mk = sMask[b * HNB + tid];
    wSs[tid] = attS[tid] * e * mk;
    aMs[tid] = attS[tid] * mk;
  } else if (tid < 40) {
    int h = tid - 20;
    float e = expf(sc[5] * dTa[h]);
    float mk = tMask[b * HNB + h];
    wTt[h] = attT[h] * e * mk;
    aMt[h] = attT[h] * mk;
  }
  __syncthreads();

  // Phase C: s_hat/t_hat -> g_s/g_t (per embedding dim)
  if (tid < 128) {
    float s = 0.f;
    for (int h = 0; h < HNB; ++h) s += aMs[h] * prS[h * 128 + tid];
    gSv[tid] = s * expf(-sc[4] * sc[6]);
  } else {
    int e = tid - 128;
    float s = 0.f;
    for (int h = 0; h < HNB; ++h) s += aMt[h] * prT[h * 128 + e];
    gTv[e] = s * expf(-sc[5] * sc[7]);
  }
  __syncthreads();

  // Phase D: gate
  if (tid == 0) {
    float ds_ = 0.f, dt_ = 0.f;
    for (int e = 0; e < 128; ++e) { ds_ += gSv[e] * gw[e]; dt_ += gTv[e] * gw[e]; }
    float ls = tanhf(ds_ + gb[0]);
    float lt = tanhf(dt_ + gb[0]);
    float mx = fmaxf(ls, lt);
    float es = expf(ls - mx), et2 = expf(lt - mx);
    float inv = 1.f / (es + et2);
    sc[8] = es * inv; sc[9] = et2 * inv;
  }
  __syncthreads();

  // Phase E: lambdas
  if (tid < NNEG) {
    float s = 0.f;
    for (int h = 0; h < HNB; ++h) s += wSs[h] * nAs[h * NNEG + tid];
    nlS[tid] = nMuS[tid] + sc[8] * s;
  } else if (tid < 2 * NNEG) {
    int nn = tid - NNEG;
    float s = 0.f;
    for (int h = 0; h < HNB; ++h) s += wTt[h] * nAt[h * NNEG + nn];
    nlT[nn] = nMuT[nn] + sc[9] * s;
  } else if (tid == 10) {
    float ss = 0.f, st = 0.f;
    for (int h = 0; h < HNB; ++h) { ss += wSs[h] * pAs[h]; st += wTt[h] * pAt[h]; }
    sc[11] = sc[2] + sc[8] * ss + sc[9] * st;       // p_lambda
  }
  __syncthreads();

  // Phase F: losses
  if (tid == 0) {
    float sig = 1.f / (1.f + expf(-sc[11] / SIGMA));
    float posv = sc[3];
    float pd = posv - 1.f;
    float pos_loss = -logf(sig + 1e-6f) * pd * pd;
    float negl = 0.f;
    for (int nn = 0; nn < NNEG; ++nn) {
      float sgs = 1.f / (1.f + expf(nlS[nn] / SIGMA));   // sigmoid(-x)
      negl += -logf(sgs + 1e-6f) * nS[nn] * nS[nn];
      float sgt = 1.f / (1.f + expf(nlT[nn] / SIGMA));
      negl += -logf(sgt + 1e-6f) * nT[nn] * nT[nn];
    }
    blockLoss[b] = pos_loss / INDEXC + negl / (INDEXC * (float)NNEG);
    posOut[b] = posv;
  }
}

// ---------------------------------------------------------------------------
// Kernel 3: deterministic fixed-order loss reduction (no float atomics)
// ---------------------------------------------------------------------------
__global__ __launch_bounds__(256) void tgsl_reduce_kernel(
    const float* __restrict__ part, float* __restrict__ out)
{
  __shared__ float s[256];
  float acc = 0.f;
  for (int i = threadIdx.x; i < BSZ; i += 256) acc += part[i];
  s[threadIdx.x] = acc;
  __syncthreads();
  if (threadIdx.x == 0) {
    float t = 0.f;
    for (int i = 0; i < 256; ++i) t += s[i];
    out[0] = t;
  }
}

extern "C" void kernel_launch(void* const* d_in, const int* in_sizes, int n_in,
                              void* d_out, int out_size, void* d_ws, size_t ws_size,
                              hipStream_t stream) {
  (void)in_sizes; (void)n_in; (void)out_size; (void)ws_size;
  const float* X       = (const float*)d_in[0];
  const int*   srcI    = (const int*)  d_in[1];
  const int*   dstI    = (const int*)  d_in[2];
  const float* eT      = (const float*)d_in[3];
  const int*   shN     = (const int*)  d_in[4];
  const int*   thN     = (const int*)  d_in[5];
  const float* sTimes  = (const float*)d_in[6];
  const float* tTimes  = (const float*)d_in[7];
  const float* sMask   = (const float*)d_in[8];
  const float* tMask   = (const float*)d_in[9];
  const int*   negSrc  = (const int*)  d_in[10];
  const int*   negDst  = (const int*)  d_in[11];
  const float* W1      = (const float*)d_in[12];
  const float* b1      = (const float*)d_in[13];
  const float* W2      = (const float*)d_in[14];
  const float* b2      = (const float*)d_in[15];
  const float* Wm      = (const float*)d_in[16];
  const float* av      = (const float*)d_in[17];
  const float* dS_p    = (const float*)d_in[18];
  const float* dT_p    = (const float*)d_in[19];
  const float* gw      = (const float*)d_in[20];
  const float* gb      = (const float*)d_in[21];

  float* out   = (float*)d_out;
  float* lossO = out;                 // [0]
  float* posO  = out + 1;             // [1 .. 4096]
  float* embO  = out + 1 + BSZ;       // node_emb, N*E floats
  float* partial = (float*)d_ws;      // 4096 per-block partial losses

  const int grid1 = (N_NODES + 127) / 128;
  tgsl_mlp_kernel<<<grid1, 256, 0, stream>>>(X, W1, b1, W2, b2, embO);
  tgsl_edge_kernel<<<BSZ, 256, 0, stream>>>(embO, srcI, dstI, eT, shN, thN,
                                            sTimes, tTimes, sMask, tMask,
                                            negSrc, negDst, Wm, av, dS_p, dT_p,
                                            gw, gb, partial, posO);
  tgsl_reduce_kernel<<<1, 256, 0, stream>>>(partial, lossO);
}
